// DecoderWithAttention_43920335569156
// MI455X (gfx1250) — matrix operand order repaired and checked by reading, compiled, not run on hardware
//
#include <hip/hip_runtime.h>
#include <math.h>

// Problem constants (from reference)
//   B=64, P=196, T=30, E=D=A=ENC=512, V=30000

typedef __bf16 bf16;
typedef __attribute__((ext_vector_type(16))) __bf16 v16bf;
typedef __attribute__((ext_vector_type(8)))  __bf16 v8bf;
typedef __attribute__((ext_vector_type(8)))  float  v8f;

// ---------------------------------------------------------------------------
// WMMA inner loop, M-blocked: one wave computes MB stacked 16x16 f32 tiles
// (a 16*MB x 16 strip) of A(MxK,bf16,row-major) times B^T(NxK,bf16,row-major).
//
// Fragment layouts per CDNA5 ISA 7.12.2 (16-bit A 16x32, B 32x16):
//   A: lane half h, row m=lane&15. Fragment = two contiguous 8-elem runs:
//      K in [k0+8h, k0+8h+8) and [k0+16+8h, k0+24+8h)  -> 2 x 16B loads.
//   B: lane n=lane&15, K in [k0+8h, k0+8h+8) and [k0+16+8h, k0+24+8h) of B^T row
//      -> 2 x 16B loads.
// All contiguous, so everything lowers to global_load_b128.
// ---------------------------------------------------------------------------
template <int MB>
__device__ inline void wmma_mb(const bf16* __restrict__ A, int lda,
                               const bf16* __restrict__ Bt, int ldbt,
                               int K, int m0, int n0, int lane, v8f acc[MB])
{
  const int half = lane >> 4;
  const int l15  = lane & 15;
  const bf16* brow = Bt + (size_t)(n0 + l15) * ldbt;
  for (int k0 = 0; k0 < K; k0 += 32) {
    v8bf b0 = *(const v8bf*)(brow + k0 + half * 8);
    v8bf b1 = *(const v8bf*)(brow + k0 + 16 + half * 8);
    v16bf b = __builtin_shufflevector(b0, b1, 0, 1, 2, 3, 4, 5, 6, 7,
                                      8, 9, 10, 11, 12, 13, 14, 15);
    if (k0 + 32 < K)
      __builtin_prefetch(brow + k0 + 32 + half * 16, 0, 0);  // global_prefetch_b8
#pragma unroll
    for (int i = 0; i < MB; ++i) {
      const bf16* arow = A + (size_t)(m0 + i * 16 + l15) * lda;
      v8bf a0 = *(const v8bf*)(arow + k0 + half * 8);
      v8bf a1 = *(const v8bf*)(arow + k0 + 16 + half * 8);
      v16bf a = __builtin_shufflevector(a0, a1, 0, 1, 2, 3, 4, 5, 6, 7,
                                        8, 9, 10, 11, 12, 13, 14, 15);
      acc[i] = __builtin_amdgcn_wmma_f32_16x16x32_bf16(
          /*neg_a=*/false, a, /*neg_b=*/false, b,
          /*c_mod=*/(short)0, acc[i], /*reuse_a=*/false, /*reuse_b=*/false);
    }
  }
}

// Generic GEMM (M multiple of 64): C(fp32) and/or Cbf(bf16) = act(A*B + bias).
// B supplied TRANSPOSED: Bt is N x K row-major.  act: 0=id, 1=tanh.
__global__ void __launch_bounds__(256)
gemm_bf16_kernel(const bf16* __restrict__ A, int lda,
                 const bf16* __restrict__ Bt, int ldbt,
                 const float* __restrict__ bias,
                 float* __restrict__ C, long long ldc,
                 bf16* __restrict__ Cbf, int ldcbf,
                 int M, int N, int K, int act)
{
  const int lane   = threadIdx.x & 31;
  const int warp   = threadIdx.x >> 5;
  const int tilesN = N >> 4;
  const int tile   = blockIdx.x * 8 + warp;            // wave-uniform
  if (tile >= (M >> 6) * tilesN) return;               // whole-wave exit
  const int m0 = (tile / tilesN) << 6;                 // 64-row strip
  const int n0 = (tile % tilesN) << 4;

  v8f z = {0.f, 0.f, 0.f, 0.f, 0.f, 0.f, 0.f, 0.f};
  v8f acc[4] = {z, z, z, z};
  wmma_mb<4>(A, lda, Bt, ldbt, K, m0, n0, lane, acc);

  const int half = lane >> 4;
  const int n    = n0 + (lane & 15);
  const float bv = bias ? bias[n] : 0.f;
#pragma unroll
  for (int i = 0; i < 4; ++i) {
#pragma unroll
    for (int r = 0; r < 8; ++r) {
      int   m = m0 + i * 16 + half * 8 + r;
      float v = acc[i][r] + bv;
      if (act == 1) v = tanhf(v);
      if (C)   C[(size_t)m * (size_t)ldc + n] = v;
      if (Cbf) Cbf[(size_t)m * ldcbf + n]     = (bf16)v;
    }
  }
}

// LSTM gates GEMM: gates(64x2048) = x(64x1024)*W_ih + h(64x512)*W_hh + biases.
// Weights supplied transposed: WihT (2048x1024), WhhT (2048x512).
__global__ void __launch_bounds__(256)
gemm_gates_kernel(const bf16* __restrict__ X,  const bf16* __restrict__ WihT,
                  const bf16* __restrict__ H,  const bf16* __restrict__ WhhT,
                  const float* __restrict__ bih, const float* __restrict__ bhh,
                  float* __restrict__ gates)
{
  const int lane = threadIdx.x & 31;
  const int warp = threadIdx.x >> 5;
  const int tile = blockIdx.x * 8 + warp;              // 128 tiles: 1 x 128
  if (tile >= 128) return;
  const int n0 = tile << 4;

  v8f z = {0.f, 0.f, 0.f, 0.f, 0.f, 0.f, 0.f, 0.f};
  v8f acc[4] = {z, z, z, z};
  wmma_mb<4>(X, 1024, WihT, 1024, 1024, 0, n0, lane, acc);
  wmma_mb<4>(H,  512, WhhT,  512,  512, 0, n0, lane, acc);

  const int half = lane >> 4;
  const int n    = n0 + (lane & 15);
  const float bv = bih[n] + bhh[n];
#pragma unroll
  for (int i = 0; i < 4; ++i)
#pragma unroll
    for (int r = 0; r < 8; ++r)
      gates[(size_t)(i * 16 + half * 8 + r) * 2048 + n] = acc[i][r] + bv;
}

// Pointwise LSTM cell update: consumes gates + c, produces c (in place) and h(bf16)
__global__ void __launch_bounds__(256)
lstm_update_kernel(const float* __restrict__ gates,
                   float* __restrict__ c, bf16* __restrict__ h_bf)
{
  int idx = blockIdx.x * blockDim.x + threadIdx.x;     // 0..32767
  int b = idx >> 9, d = idx & 511;
  const float* g = gates + (size_t)b * 2048;
  float ig = g[d], fg = g[512 + d], gg = g[1024 + d], og = g[1536 + d];
  float si = 1.f / (1.f + __expf(-ig));
  float sf = 1.f / (1.f + __expf(-fg));
  float so = 1.f / (1.f + __expf(-og));
  float c2 = sf * c[idx] + si * tanhf(gg);
  float h2 = so * tanhf(c2);
  c[idx]    = c2;
  h_bf[idx] = (bf16)h2;
}

// scores[b,p] = sum_a tanh(enc_proj[b,p,a] + dec_proj[b,a]) * wf[a] + bf
// One wave per (b,p): 64*196 = 12544 waves = 1568 blocks of 8 waves.
__global__ void __launch_bounds__(256)
attn_scores_kernel(const float* __restrict__ enc_proj,
                   const float* __restrict__ dec_proj,
                   const float* __restrict__ wf,
                   const float* __restrict__ bf_s,
                   float* __restrict__ scores)
{
  int w = blockIdx.x * 8 + (threadIdx.x >> 5);
  if (w >= 64 * 196) return;
  int lane = threadIdx.x & 31;
  int b = w / 196, p = w % 196;
  const float* ep = enc_proj + ((size_t)b * 196 + p) * 512;
  const float* dp = dec_proj + (size_t)b * 512;
  float s = 0.f;
#pragma unroll 4
  for (int a = lane; a < 512; a += 32)
    s += tanhf(ep[a] + dp[a]) * wf[a];
#pragma unroll
  for (int off = 16; off > 0; off >>= 1)
    s += __shfl_xor(s, off, 32);                       // wave32 reduction
  if (lane == 0) scores[(size_t)b * 196 + p] = s + bf_s[0];
}

// softmax over P=196 per batch row
__global__ void __launch_bounds__(256)
softmax_kernel(const float* __restrict__ scores, float* __restrict__ alpha)
{
  __shared__ float sm[256];
  int b = blockIdx.x, tid = threadIdx.x;
  float v = (tid < 196) ? scores[(size_t)b * 196 + tid] : -3.4e38f;
  sm[tid] = v; __syncthreads();
  for (int s = 128; s > 0; s >>= 1) {
    if (tid < s) sm[tid] = fmaxf(sm[tid], sm[tid + s]);
    __syncthreads();
  }
  float mx = sm[0]; __syncthreads();
  float e = (tid < 196) ? __expf(v - mx) : 0.f;
  sm[tid] = e; __syncthreads();
  for (int s = 128; s > 0; s >>= 1) {
    if (tid < s) sm[tid] += sm[tid + s];
    __syncthreads();
  }
  float inv = 1.f / sm[0];
  if (tid < 196) alpha[(size_t)b * 196 + tid] = e * inv;
}

// ctx[b,d] = sigmoid(beta_lin[b,d]) * sum_p alpha[b,p]*feats[b,p,d]
// also builds x_bf16 = [ emb_table[captions[b,t-1]] , ctx ]  (64 x 1024)
__global__ void __launch_bounds__(256)
ctx_build_x_kernel(const float* __restrict__ alpha,
                   const float* __restrict__ feats,
                   const float* __restrict__ beta_lin,
                   const int*   __restrict__ captions,
                   const float* __restrict__ emb_table,
                   int t, bf16* __restrict__ x_bf)
{
  __shared__ float al[196];
  int b = blockIdx.x, tid = threadIdx.x;
  if (tid < 196) al[tid] = alpha[(size_t)b * 196 + tid];
  __syncthreads();
  int tok = captions[b * 30 + (t - 1)];
#pragma unroll
  for (int j = 0; j < 2; ++j) {
    int d = tid + j * 256;
    const float* fp = feats + (size_t)b * 196 * 512 + d;
    float s = 0.f;
    for (int p = 0; p < 196; ++p) s += al[p] * fp[(size_t)p * 512];
    float gate = 1.f / (1.f + __expf(-beta_lin[(size_t)b * 512 + d]));
    x_bf[(size_t)b * 1024 + 512 + d] = (bf16)(gate * s);
    x_bf[(size_t)b * 1024 + d]       = (bf16)emb_table[(size_t)tok * 512 + d];
  }
}

// Plain f32 -> bf16 conversion (activations: layout preserved)
__global__ void f32_to_bf16_kernel(const float* __restrict__ src,
                                   bf16* __restrict__ dst, int n)
{
  int i = blockIdx.x * blockDim.x + threadIdx.x;
  if (i < n) dst[i] = (bf16)src[i];
}

// Transposing f32 -> bf16 conversion for weights: src is KxN, dst is NxK.
// 16x16 LDS tile; all K,N here are multiples of 16 (incl. 30000 = 1875*16).
__global__ void __launch_bounds__(256)
transpose_cvt_kernel(const float* __restrict__ src, bf16* __restrict__ dst,
                     int K, int N)
{
  __shared__ bf16 tile[16][17];
  const int n0 = blockIdx.x * 16, k0 = blockIdx.y * 16;
  const int tx = threadIdx.x & 15, ty = threadIdx.x >> 4;
  tile[ty][tx] = (bf16)src[(size_t)(k0 + ty) * N + (n0 + tx)];
  __syncthreads();
  dst[(size_t)(n0 + ty) * K + (k0 + tx)] = tile[tx][ty];
}

// out[:, 0, :] = 0
__global__ void zero_t0_kernel(float* __restrict__ out)
{
  int idx = blockIdx.x * blockDim.x + threadIdx.x;
  if (idx < 64 * 30000) {
    int b = idx / 30000, n = idx % 30000;
    out[(size_t)b * 900000 + n] = 0.f;
  }
}

// ---------------------------------------------------------------------------
extern "C" void kernel_launch(void* const* d_in, const int* in_sizes, int n_in,
                              void* d_out, int out_size, void* d_ws, size_t ws_size,
                              hipStream_t stream)
{
  (void)in_sizes; (void)n_in; (void)out_size; (void)ws_size;
  const float* feats     = (const float*)d_in[0];
  const float* pooled    = (const float*)d_in[1];
  const int*   captions  = (const int*)  d_in[2];
  const float* We_att    = (const float*)d_in[3];
  const float* be_att    = (const float*)d_in[4];
  const float* Wd_att    = (const float*)d_in[5];
  const float* bd_att    = (const float*)d_in[6];
  const float* wf_att    = (const float*)d_in[7];
  const float* bf_att    = (const float*)d_in[8];
  const float* emb_table = (const float*)d_in[9];
  const float* W_ih      = (const float*)d_in[10];
  const float* W_hh      = (const float*)d_in[11];
  const float* b_ih      = (const float*)d_in[12];
  const float* b_hh      = (const float*)d_in[13];
  const float* W_init_h  = (const float*)d_in[14];
  const float* b_init_h  = (const float*)d_in[15];
  const float* W_init_c  = (const float*)d_in[16];
  const float* b_init_c  = (const float*)d_in[17];
  const float* W_fc      = (const float*)d_in[18];
  const float* b_fc      = (const float*)d_in[19];
  const float* W_beta    = (const float*)d_in[20];
  const float* b_beta    = (const float*)d_in[21];
  float* out = (float*)d_out;

  char* base = (char*)d_ws;
  size_t off = 0;
  auto alloc = [&](size_t bytes) -> char* {
    char* p = base + off;
    off += (bytes + 255) & ~(size_t)255;
    return p;
  };

  // Transposed bf16 weights (N x K, row-major)
  bf16*  wfcT    = (bf16*) alloc(30000ULL * 512 * 2);
  bf16*  wihT    = (bf16*) alloc(2048ULL * 1024 * 2);
  bf16*  whhT    = (bf16*) alloc(2048ULL * 512 * 2);
  bf16*  weattT  = (bf16*) alloc(512ULL * 512 * 2);
  bf16*  wdattT  = (bf16*) alloc(512ULL * 512 * 2);
  bf16*  wbetaT  = (bf16*) alloc(512ULL * 512 * 2);
  bf16*  winithT = (bf16*) alloc(512ULL * 512 * 2);
  bf16*  winitcT = (bf16*) alloc(512ULL * 512 * 2);
  // bf16 activations (row-major)
  bf16*  feats_bf  = (bf16*) alloc(64ULL * 196 * 512 * 2);
  bf16*  pooled_bf = (bf16*) alloc(64ULL * 512 * 2);
  // fp32 state / scratch
  float* enc_proj  = (float*)alloc(64ULL * 196 * 512 * 4);
  float* c_state   = (float*)alloc(64ULL * 512 * 4);
  bf16*  h_bf      = (bf16*) alloc(64ULL * 512 * 2);
  float* dec_proj  = (float*)alloc(64ULL * 512 * 4);
  float* beta_lin  = (float*)alloc(64ULL * 512 * 4);
  float* scores    = (float*)alloc(64ULL * 196 * 4);
  float* alpha     = (float*)alloc(64ULL * 196 * 4);
  bf16*  x_bf      = (bf16*) alloc(64ULL * 1024 * 2);
  float* gates     = (float*)alloc(64ULL * 2048 * 4);

  // Weight conversions (transpose): src K x N -> dst N x K
  auto tcvt = [&](const float* s, bf16* d, int K, int N) {
    dim3 g(N / 16, K / 16);
    transpose_cvt_kernel<<<g, 256, 0, stream>>>(s, d, K, N);
  };
  tcvt(W_fc,     wfcT,    512,  30000);
  tcvt(W_ih,     wihT,    1024, 2048);
  tcvt(W_hh,     whhT,    512,  2048);
  tcvt(We_att,   weattT,  512,  512);
  tcvt(Wd_att,   wdattT,  512,  512);
  tcvt(W_beta,   wbetaT,  512,  512);
  tcvt(W_init_h, winithT, 512,  512);
  tcvt(W_init_c, winitcT, 512,  512);

  // Activation conversions (no transpose)
  auto cvt = [&](const float* s, bf16* d, size_t n) {
    f32_to_bf16_kernel<<<(unsigned)((n + 255) / 256), 256, 0, stream>>>(s, d, (int)n);
  };
  cvt(feats,  feats_bf,  64ULL * 196 * 512);
  cvt(pooled, pooled_bf, 64ULL * 512);

  auto gemm = [&](const bf16* A, int lda, const bf16* Bt, int ldbt,
                  const float* bias, float* C, long long ldc,
                  bf16* Cbf, int ldcbf, int M, int N, int K, int act) {
    int tiles = (M / 64) * (N / 16);
    gemm_bf16_kernel<<<(tiles + 7) / 8, 256, 0, stream>>>(
        A, lda, Bt, ldbt, bias, C, ldc, Cbf, ldcbf, M, N, K, act);
  };

  // Precompute: enc_proj = feats @ We_att + be_att   (12544 x 512, K=512)
  gemm(feats_bf, 512, weattT, 512, be_att, enc_proj, 512, nullptr, 0,
       12544, 512, 512, 0);
  // h0 = tanh(pooled @ W_init_h + b) -> bf16 ; c0 = tanh(pooled @ W_init_c + b) -> f32
  gemm(pooled_bf, 512, winithT, 512, b_init_h, nullptr, 0, h_bf, 512,
       64, 512, 512, 1);
  gemm(pooled_bf, 512, winitcT, 512, b_init_c, c_state, 512, nullptr, 0,
       64, 512, 512, 1);
  // out[:,0,:] = 0
  zero_t0_kernel<<<(64 * 30000 + 255) / 256, 256, 0, stream>>>(out);

  for (int t = 1; t < 30; ++t) {
    // attention projections from current h
    gemm(h_bf, 512, wdattT, 512, bd_att, dec_proj, 512, nullptr, 0,
         64, 512, 512, 0);
    gemm(h_bf, 512, wbetaT, 512, b_beta, beta_lin, 512, nullptr, 0,
         64, 512, 512, 0);
    attn_scores_kernel<<<1568, 256, 0, stream>>>(enc_proj, dec_proj, wf_att,
                                                 bf_att, scores);
    softmax_kernel<<<64, 256, 0, stream>>>(scores, alpha);
    ctx_build_x_kernel<<<64, 256, 0, stream>>>(alpha, feats, beta_lin,
                                               captions, emb_table, t, x_bf);
    // LSTM gates + cell update
    gemm_gates_kernel<<<16, 256, 0, stream>>>(x_bf, wihT, h_bf, whhT,
                                              b_ih, b_hh, gates);
    lstm_update_kernel<<<128, 256, 0, stream>>>(gates, c_state, h_bf);
    // FC head: out[:, t, :] = h2 @ W_fc + b_fc   (64 x 30000, K=512)
    gemm(h_bf, 512, wfcT, 512, b_fc, out + (size_t)t * 30000, 900000LL,
         nullptr, 0, 64, 30000, 512, 0);
  }
}
